// FODFReorientation_8375186227649
// MI455X (gfx1250) — matrix operand order repaired and checked by reading, not compiled, not run on hardware
//
#include <hip/hip_runtime.h>
#include <hip/hip_bf16.h>

// ---------------------------------------------------------------------------
// FODF reorientation for MI455X (gfx1250, wave32).
//   Kernel 1: W[V=64, N] = M_p(64x45) @ img^T(45xN) via v_wmma_f32_16x16x32_f16
//             with 3-term hi/lo f16 split for fp32-class accuracy.
//   Kernel 2: per-voxel 3x3 inverse, reorient, degree-8 even SH (registers),
//             modulation, LDS reduction over 64 directions -> 45 coefficients.
// ---------------------------------------------------------------------------

typedef __attribute__((ext_vector_type(16))) _Float16 v16h;
typedef __attribute__((ext_vector_type(8)))  float    v8f;

#define NC 45   // SH coefficients (lmax=8, even orders)
#define NV 64   // PSF directions

// ------------------------- compile-time SH constants ------------------------
constexpr double csqrt_(double x) {
  double g = x > 1.0 ? x : 1.0;
  for (int i = 0; i < 100; ++i) g = 0.5 * (g + x / g);
  return g;
}
constexpr double cdfact_(int n) { double o = 1.0; for (int k = n; k > 1; k -= 2) o *= (double)k; return o; }
constexpr double cfact_(int n)  { double o = 1.0; for (int k = 2; k <= n; ++k)  o *= (double)k; return o; }

struct CT25 { float v[25]; };
constexpr CT25 make_ct() {
  CT25 t{};
  const double PI  = 3.14159265358979323846;
  const double SQ2 = csqrt_(2.0);
  for (int l = 0; l <= 8; l += 2)
    for (int am = 0; am <= l; ++am) {
      double K   = csqrt_((2.0 * l + 1.0) / (4.0 * PI) * cfact_(l - am) / cfact_(l + am));
      double sgn = (am & 1) ? -1.0 : 1.0;        // Tournier sign
      double s2  = (am > 0) ? SQ2 : 1.0;
      t.v[(l / 2) * (l / 2) + am] = (float)(sgn * s2 * K);
    }
  return t;
}
constexpr CT25 CT = make_ct();

struct Q9 { float v[9]; };
constexpr Q9 make_q() {
  Q9 q{};
  for (int m = 0; m <= 8; ++m) {
    double s = (m & 1) ? -1.0 : 1.0;
    q.v[m] = (float)(s * cdfact_(2 * m - 1));    // Q(m,m) = (-1)^m (2m-1)!!
  }
  return q;
}
constexpr Q9 QMM = make_q();

// --------------------- Kernel 1: W = M_p @ img^T (WMMA) ---------------------
// Block = 128 threads (4 waves), tile = 16 voxels. Wave w owns direction rows
// 16w..16w+15. K = 45 padded to 64 -> two K=32 WMMA steps, 3 hi/lo terms each.
__global__ __launch_bounds__(128)
void fodf_wmma_weights(const float* __restrict__ img,  // [N,45]
                       const float* __restrict__ Mp,   // [64,45]
                       float* __restrict__ W,          // [N,64]
                       int N) {
  const int v0  = blockIdx.x * 16;   // first voxel of this tile
  const int tid = threadIdx.x;

  __shared__ _Float16 s_mp_h[64 * 64];
  __shared__ _Float16 s_mp_l[64 * 64];
  __shared__ _Float16 s_im_h[16 * 64];
  __shared__ _Float16 s_im_l[16 * 64];

  // Stage M_p (hi/lo f16 split, K padded to 64)
  for (int i = tid; i < 64 * 64; i += 128) {
    int r = i >> 6, k = i & 63;
    float x = (k < NC) ? Mp[r * NC + k] : 0.f;
    _Float16 h = (_Float16)x;
    s_mp_h[i] = h;
    s_mp_l[i] = (_Float16)(x - (float)h);
  }
  // Stage 16-voxel image tile (hi/lo f16 split)
  for (int i = tid; i < 16 * 64; i += 128) {
    int r = i >> 6, k = i & 63;
    int vox = v0 + r;
    float x = (k < NC && vox < N) ? img[(size_t)vox * NC + k] : 0.f;
    _Float16 h = (_Float16)x;
    s_im_h[i] = h;
    s_im_l[i] = (_Float16)(x - (float)h);
  }
  __syncthreads();

  const int wave    = tid >> 5;          // 0..3
  const int lane    = tid & 31;
  const int halfsel = (lane >= 16) ? 1 : 0;
  const int lrow    = lane & 15;

  v8f acc = {};
#pragma unroll
  for (int s = 0; s < 2; ++s) {
    const int K0 = s * 32;
    v16h ah, al, bh, bl;
#pragma unroll
    for (int h = 0; h < 16; ++h) {
      // 16-bit A-matrix 16x32 layout: lane half selects K+8/K+24 interleave
      int ka = K0 + ((h >> 3) << 4) + (halfsel << 3) + (h & 7);
      int ia = (wave * 16 + lrow) * 64 + ka;
      ah[h] = s_mp_h[ia];
      al[h] = s_mp_l[ia];
      // 16-bit B-matrix 32x16 layout: same K+8/K+24 interleave as A
      int kb = K0 + ((h >> 3) << 4) + (halfsel << 3) + (h & 7);
      int ib = lrow * 64 + kb;
      bh[h] = s_im_h[ib];
      bl[h] = s_im_l[ib];
    }
    // D = Ah*Bh + Ah*Bl + Al*Bh + C  (drops ~2^-22 lo*lo term)
    acc = __builtin_amdgcn_wmma_f32_16x16x32_f16(false, ah, false, bh, (short)0, acc, false, false);
    acc = __builtin_amdgcn_wmma_f32_16x16x32_f16(false, ah, false, bl, (short)0, acc, false, false);
    acc = __builtin_amdgcn_wmma_f32_16x16x32_f16(false, al, false, bh, (short)0, acc, false, false);
  }

  // C/D layout: lane%16 = column (voxel), VGPR r -> row r (+8 for upper lanes)
#pragma unroll
  for (int r = 0; r < 8; ++r) {
    int vec = wave * 16 + r + (halfsel << 3);
    int vox = v0 + lrow;
    if (vox < N) W[(size_t)vox * NV + vec] = acc[r];
  }
}

// --------- Kernel 2: reorient + SH + modulate + reduce (1 block/voxel) ------
// EMIT writes the (l, +/-m) output slots; block offset = l(l-1)/2 for even l.
#define EMIT(l, m, q)                                        \
  do {                                                       \
    int   bo_ = (l) * ((l) - 1) / 2;                         \
    float c_  = CT.v[((l) >> 1) * ((l) >> 1) + (m)];         \
    float t_  = c_ * (q);                                    \
    if ((m) == 0) {                                          \
      sh[bo_ + (l)] = t_;                                    \
    } else {                                                 \
      sh[bo_ + (l) + (m)] = t_ * Am[(m)];                    \
      sh[bo_ + (l) - (m)] = t_ * Bm[(m)];                    \
    }                                                        \
  } while (0)

__global__ __launch_bounds__(NV)
void fodf_reorient(const float* __restrict__ img,     // [N,45] (fallback path)
                   const float* __restrict__ affine,  // [N,3,3]
                   const float* __restrict__ psf,     // [64,3]
                   const float* __restrict__ Mp,      // [64,45] (fallback path)
                   const float* __restrict__ W,       // [N,64] or nullptr
                   float* __restrict__ out,           // [N,45]
                   int N) {
  const int n = blockIdx.x;
  const int v = threadIdx.x;  // direction index 0..63
  if (n >= N) return;

  // ---- per-voxel 3x3 inverse + det (redundant per thread; broadcast loads)
  const float* A = affine + (size_t)n * 9;
  float a00 = A[0], a01 = A[1], a02 = A[2];
  float a10 = A[3], a11 = A[4], a12 = A[5];
  float a20 = A[6], a21 = A[7], a22 = A[8];
  float c00 = a11 * a22 - a12 * a21;
  float c01 = a12 * a20 - a10 * a22;
  float c02 = a10 * a21 - a11 * a20;
  float det = a00 * c00 + a01 * c01 + a02 * c02;
  float id  = 1.f / det;
  float i00 = c00 * id, i01 = (a02 * a21 - a01 * a22) * id, i02 = (a01 * a12 - a02 * a11) * id;
  float i10 = c01 * id, i11 = (a00 * a22 - a02 * a20) * id, i12 = (a02 * a10 - a00 * a12) * id;
  float i20 = c02 * id, i21 = (a01 * a20 - a00 * a21) * id, i22 = (a00 * a11 - a01 * a10) * id;

  // ---- reorient direction v
  float vx = psf[v * 3 + 0], vy = psf[v * 3 + 1], vz = psf[v * 3 + 2];
  float hx = i00 * vx + i01 * vy + i02 * vz;
  float hy = i10 * vx + i11 * vy + i12 * vz;
  float hz = i20 * vx + i21 * vy + i22 * vz;
  float r2 = hx * hx + hy * hy + hz * hz;
  float rn = rsqrtf(r2 + 1e-12f);
  // reference permutes xyz -> yzx before SH
  float px = hy * rn, py = hz * rn, pz = hx * rn;

  // ---- PSF weight: from WMMA kernel, or recompute if no workspace
  float wv;
  if (W) {
    wv = W[(size_t)n * NV + v];
  } else {
    wv = 0.f;
    const float* ic = img + (size_t)n * NC;
    const float* mr = Mp + v * NC;
#pragma unroll
    for (int c = 0; c < NC; ++c) wv = fmaf(mr[c], ic[c], wv);
  }
  // modulation: |A^{-1} v| / det(A^{-1}) = |vi_hat| * det(A)
  float s = sqrtf(r2) * det * wv;

  // ---- azimuthal recurrence A_m = rho^m cos(m phi), B_m = rho^m sin(m phi)
  float Am[9], Bm[9];
  Am[0] = 1.f; Bm[0] = 0.f;
#pragma unroll
  for (int m = 1; m <= 8; ++m) {
    Am[m] = px * Am[m - 1] - py * Bm[m - 1];
    Bm[m] = px * Bm[m - 1] + py * Am[m - 1];
  }

  // ---- associated Legendre Q(l,m) = P_l^m / sin^m(theta), polynomial in pz
  float sh[NC];
#pragma unroll
  for (int m = 0; m <= 8; ++m) {
    float q2 = QMM.v[m];                       // Q(m,m)
    float q1 = (float)(2 * m + 1) * q2 * pz;   // Q(m+1,m)
    if ((m & 1) == 0) EMIT(m, m, q2);
    if (m < 8 && (((m + 1) & 1) == 0)) EMIT(m + 1, m, q1);
#pragma unroll
    for (int l = m + 2; l <= 8; ++l) {
      float q = ((float)(2 * l - 1) * pz * q1 - (float)(l + m - 1) * q2) *
                (1.f / (float)(l - m));
      q2 = q1; q1 = q;
      if ((l & 1) == 0) EMIT(l, m, q);
    }
  }

  // ---- scale row by w_mod and reduce over the 64 directions via LDS
  __shared__ float ls[NV][NC + 1];   // pad to dodge bank conflicts
#pragma unroll
  for (int c = 0; c < NC; ++c) ls[v][c] = sh[c] * s;
  __syncthreads();

  if (v < NC) {
    float acc = 0.f;
#pragma unroll
    for (int j = 0; j < NV; ++j) acc += ls[j][v];
    out[(size_t)n * NC + v] = acc;
  }
}

// ------------------------------- launcher -----------------------------------
extern "C" void kernel_launch(void* const* d_in, const int* in_sizes, int n_in,
                              void* d_out, int out_size, void* d_ws, size_t ws_size,
                              hipStream_t stream) {
  const float* img = (const float*)d_in[0];  // [N,45]
  const float* aff = (const float*)d_in[1];  // [N,9]
  const float* psf = (const float*)d_in[2];  // [64,3]
  const float* Mp  = (const float*)d_in[3];  // [64,45]
  float* out = (float*)d_out;

  const int N = in_sizes[0] / NC;            // 32768 voxels
  const size_t w_bytes = (size_t)N * NV * sizeof(float);

  float* W = nullptr;
  if (d_ws && ws_size >= w_bytes) {
    W = (float*)d_ws;
    const int tiles = (N + 15) / 16;
    fodf_wmma_weights<<<tiles, 128, 0, stream>>>(img, Mp, W, N);
  }
  fodf_reorient<<<N, NV, 0, stream>>>(img, aff, psf, Mp, W, out, N);
}
